// HETA_30657476559617
// MI455X (gfx1250) — compile-verified, matrix operand order checked by reading, not run on hardware
//
#include <hip/hip_runtime.h>
#include <hip/hip_bf16.h>

#define NNODES 50000
#define NEDGE  500000
#define DIM    128
#define LPAD   136   // LDS row stride in halfwords (128 + 8 pad)

typedef __attribute__((ext_vector_type(16))) _Float16 v16h;
typedef __attribute__((ext_vector_type(8)))  _Float16 v8h;
typedef __attribute__((ext_vector_type(8)))  float    v8f;

union frag16 { v16h v; v8h h[2]; };

// ---------------------------------------------------------------------------
// Order-preserving float<->int encoding so segment-max can use atomicMax(int).
// ---------------------------------------------------------------------------
__device__ __forceinline__ int enc_f(float f) {
    int i = __float_as_int(f);
    return i >= 0 ? i : (i ^ 0x7FFFFFFF);
}
__device__ __forceinline__ float dec_f(int i) {
    return __int_as_float(i >= 0 ? i : (i ^ 0x7FFFFFFF));
}

// ---------------------------------------------------------------------------
// Init: out = bias sums; maxenc = enc(-inf); ssum = 0.
// ---------------------------------------------------------------------------
__global__ __launch_bounds__(256)
void init_kernel(float* __restrict__ out,
                 const float* __restrict__ b_dd, const float* __restrict__ b_tt,
                 const float* __restrict__ b_dt,
                 int* __restrict__ maxenc, float* __restrict__ ssum)
{
    size_t idx = (size_t)blockIdx.x * 256 + threadIdx.x;
    size_t total = 2ul * NNODES * DIM;
    if (idx < total) {
        int col = (int)(idx & (DIM - 1));
        float base = (idx < (size_t)NNODES * DIM) ? b_dd[col] : b_tt[col];
        out[idx] = base + b_dt[col];
    }
    if (idx < 4ul * NNODES) {
        maxenc[idx] = (int)0x807FFFFFu;   // enc(-inf)
        ssum[idx]   = 0.0f;
    }
}

// ---------------------------------------------------------------------------
// WMMA GEMM: F[nrows x 128] = X[nrows x 128] @ W[128 x 128], f32 in/out.
// Hi/lo f16 split (3 WMMAs/tile) recovers ~f32 accuracy; GEMM is <1% of
// runtime so we spend FLOPs, not bandwidth, on precision.
// Fragment layout (16-bit A/B, doc + mirrored): lane = row(A)/col(B) mod 16,
// half = lane/16; a lane's 16 K-values are the two contiguous runs
// [kbase+half*8, +8) and [kbase+half*8+16, +8)  -> b128 loads, no packing.
// Block = 256 thr = 8 waves; each wave owns a 16-row strip x all 8 N-tiles.
// ---------------------------------------------------------------------------
__global__ __launch_bounds__(256)
void gemm_wmma(const float* __restrict__ X, const float* __restrict__ W,
               float* __restrict__ F, int nrows)
{
    // W transposed in LDS: wT[n*LPAD + k], hi/lo f16 split.
    __shared__ __attribute__((aligned(16))) _Float16 wT_hi[DIM * LPAD];
    __shared__ __attribute__((aligned(16))) _Float16 wT_lo[DIM * LPAD];

    // Stage W (coalesced float4 global reads, scattered u16 LDS writes).
    const float4* W4 = reinterpret_cast<const float4*>(W);
    for (int i = threadIdx.x; i < (DIM * DIM) / 4; i += 256) {
        int flat = i * 4;
        int k = flat >> 7;         // row of W
        int n = flat & (DIM - 1);  // col of W
        float4 f = W4[i];
        float vv[4] = {f.x, f.y, f.z, f.w};
        #pragma unroll
        for (int c = 0; c < 4; ++c) {
            _Float16 hi = (_Float16)vv[c];
            wT_hi[(n + c) * LPAD + k] = hi;
            wT_lo[(n + c) * LPAD + k] = (_Float16)(vv[c] - (float)hi);
        }
    }
    __syncthreads();

    const int wave = threadIdx.x >> 5;
    const int lane = threadIdx.x & 31;
    const int half = lane >> 4;
    const int l16  = lane & 15;
    const int arow = blockIdx.x * 128 + wave * 16 + l16;  // A row for this lane
    const bool rv  = arow < nrows;

    v8f acc[8] = {};
    const float4 z4 = {0.f, 0.f, 0.f, 0.f};

    for (int ks = 0; ks < 4; ++ks) {
        const int kbase = ks * 32;

        // ---- A fragment: 4x global_load_b128, then f32 -> hi/lo f16 ----
        const float4* xp4 = reinterpret_cast<const float4*>(
            X + (size_t)arow * DIM + kbase);
        const int i0 = 2 * half;                // k offset half*8, in float4s
        float4 g0 = rv ? xp4[i0]     : z4;      // k: half*8 + 0..3
        float4 g1 = rv ? xp4[i0 + 1] : z4;      // k: half*8 + 4..7
        float4 g2 = rv ? xp4[i0 + 4] : z4;      // k: half*8 + 16..19
        float4 g3 = rv ? xp4[i0 + 5] : z4;      // k: half*8 + 20..23
        float xs[16] = {g0.x, g0.y, g0.z, g0.w, g1.x, g1.y, g1.z, g1.w,
                        g2.x, g2.y, g2.z, g2.w, g3.x, g3.y, g3.z, g3.w};
        frag16 a_hi, a_lo;
        #pragma unroll
        for (int j = 0; j < 16; ++j) {
            _Float16 h = (_Float16)xs[j];
            a_hi.h[j >> 3][j & 7] = h;
            a_lo.h[j >> 3][j & 7] = (_Float16)(xs[j] - (float)h);
        }

        #pragma unroll
        for (int nt = 0; nt < 8; ++nt) {
            const int col = nt * 16 + l16;
            const int base_hw = col * LPAD + kbase + half * 8;  // 16B-aligned
            // ---- B fragments: 2x ds_load_b128 each ----
            frag16 b_hi, b_lo;
            b_hi.h[0] = *reinterpret_cast<const v8h*>(wT_hi + base_hw);
            b_hi.h[1] = *reinterpret_cast<const v8h*>(wT_hi + base_hw + 16);
            b_lo.h[0] = *reinterpret_cast<const v8h*>(wT_lo + base_hw);
            b_lo.h[1] = *reinterpret_cast<const v8h*>(wT_lo + base_hw + 16);

            acc[nt] = __builtin_amdgcn_wmma_f32_16x16x32_f16(
                false, a_hi.v, false, b_hi.v, (short)0, acc[nt], false, false);
            acc[nt] = __builtin_amdgcn_wmma_f32_16x16x32_f16(
                false, a_lo.v, false, b_hi.v, (short)0, acc[nt], false, false);
            acc[nt] = __builtin_amdgcn_wmma_f32_16x16x32_f16(
                false, a_hi.v, false, b_lo.v, (short)0, acc[nt], false, false);
        }
    }

    // C/D layout: VGPR r -> row (8*half + r), col = lane%16 within the tile.
    #pragma unroll
    for (int nt = 0; nt < 8; ++nt) {
        const int col = nt * 16 + l16;
        #pragma unroll
        for (int r = 0; r < 8; ++r) {
            int orow = blockIdx.x * 128 + wave * 16 + half * 8 + r;
            if (orow < nrows) F[(size_t)orow * DIM + col] = acc[nt][r];
        }
    }
}

// ---------------------------------------------------------------------------
// el/er attention projections: one wave per node, float4/lane + shfl reduce.
// F index 0=x_d@W_dd 1=x_t@W_tt 2=x_d@W_dt 3=x_t@W_dt.
// ---------------------------------------------------------------------------
__global__ __launch_bounds__(256)
void dots_kernel(const float* __restrict__ Fbase,
                 const float* __restrict__ al_dd, const float* __restrict__ ar_dd,
                 const float* __restrict__ al_tt, const float* __restrict__ ar_tt,
                 const float* __restrict__ al_dt, const float* __restrict__ ar_dt,
                 float* __restrict__ el_arr, float* __restrict__ er_arr)
{
    const int fidx = blockIdx.y;
    const float* al; const float* ar;
    switch (fidx) {
        case 0:  al = al_dd; ar = ar_dd; break;
        case 1:  al = al_tt; ar = ar_tt; break;
        default: al = al_dt; ar = ar_dt; break;  // both dt projections
    }
    const int wave = threadIdx.x >> 5;
    const int lane = threadIdx.x & 31;
    const int node = blockIdx.x * 8 + wave;
    if (node >= NNODES) return;

    const float4* fr = reinterpret_cast<const float4*>(
        Fbase + (size_t)fidx * NNODES * DIM + (size_t)node * DIM);
    const float4 f = fr[lane];
    const float4 a = reinterpret_cast<const float4*>(al)[lane];
    const float4 b = reinterpret_cast<const float4*>(ar)[lane];
    float sl = f.x * a.x + f.y * a.y + f.z * a.z + f.w * a.w;
    float sr = f.x * b.x + f.y * b.y + f.z * b.z + f.w * b.w;
    #pragma unroll
    for (int o = 16; o > 0; o >>= 1) {
        sl += __shfl_xor(sl, o, 32);
        sr += __shfl_xor(sr, o, 32);
    }
    if (lane == 0) {
        el_arr[fidx * NNODES + node] = sl;
        er_arr[fidx * NNODES + node] = sr;
    }
}

// ---------------------------------------------------------------------------
// Edge logits + segment max via order-preserving atomicMax(int).
// rel 0=dd 1=tt 2=dt 3=td (dt/td share weights but swap el/er sources).
// ---------------------------------------------------------------------------
__global__ __launch_bounds__(256)
void logits_kernel(const float* __restrict__ el_arr, const float* __restrict__ er_arr,
                   const int* __restrict__ src_dd, const int* __restrict__ dst_dd,
                   const int* __restrict__ src_tt, const int* __restrict__ dst_tt,
                   const int* __restrict__ src_dt, const int* __restrict__ dst_dt,
                   const int* __restrict__ src_td, const int* __restrict__ dst_td,
                   float* __restrict__ elog, int* __restrict__ maxenc)
{
    const int rel = blockIdx.y;
    const int* src; const int* dst; const float* el; const float* er;
    switch (rel) {
        case 0:  src = src_dd; dst = dst_dd; el = el_arr;              er = er_arr;              break;
        case 1:  src = src_tt; dst = dst_tt; el = el_arr + NNODES;     er = er_arr + NNODES;     break;
        case 2:  src = src_dt; dst = dst_dt; el = el_arr + 2 * NNODES; er = er_arr + 3 * NNODES; break;
        default: src = src_td; dst = dst_td; el = el_arr + 3 * NNODES; er = er_arr + 2 * NNODES; break;
    }
    const size_t e = (size_t)blockIdx.x * 256 + threadIdx.x;
    if (e >= NEDGE) return;
    float v = el[src[e]] + er[dst[e]];
    v = (v >= 0.0f) ? v : 0.2f * v;            // leaky_relu, slope 0.2
    elog[(size_t)rel * NEDGE + e] = v;
    atomicMax(&maxenc[rel * NNODES + dst[e]], enc_f(v));
}

// ---------------------------------------------------------------------------
// exp(e - m[dst]) + segment sum via atomicAdd(f32). elog overwritten in place.
// ---------------------------------------------------------------------------
__global__ __launch_bounds__(256)
void expsum_kernel(const int* __restrict__ dst_dd, const int* __restrict__ dst_tt,
                   const int* __restrict__ dst_dt, const int* __restrict__ dst_td,
                   float* __restrict__ elog, const int* __restrict__ maxenc,
                   float* __restrict__ ssum)
{
    const int rel = blockIdx.y;
    const int* dst;
    switch (rel) {
        case 0:  dst = dst_dd; break;
        case 1:  dst = dst_tt; break;
        case 2:  dst = dst_dt; break;
        default: dst = dst_td; break;
    }
    const size_t e = (size_t)blockIdx.x * 256 + threadIdx.x;
    if (e >= NEDGE) return;
    const int d = dst[e];
    const float m  = dec_f(maxenc[rel * NNODES + d]);
    const float ex = expf(elog[(size_t)rel * NEDGE + e] - m);
    elog[(size_t)rel * NEDGE + e] = ex;
    atomicAdd(&ssum[rel * NNODES + d], ex);
}

// ---------------------------------------------------------------------------
// Aggregation: one wave per edge; float4 gather of fs[src], alpha-scaled
// atomic scatter into out[dst]. All traffic is L2-resident (192 MB).
// ---------------------------------------------------------------------------
__global__ __launch_bounds__(256)
void aggregate_kernel(const float* __restrict__ Fbase, const float* __restrict__ elog,
                      const float* __restrict__ ssum,
                      const int* __restrict__ src_dd, const int* __restrict__ dst_dd,
                      const int* __restrict__ src_tt, const int* __restrict__ dst_tt,
                      const int* __restrict__ src_dt, const int* __restrict__ dst_dt,
                      const int* __restrict__ src_td, const int* __restrict__ dst_td,
                      float* __restrict__ out)
{
    const int rel = blockIdx.y;
    const int* src; const int* dst; const float* Fsrc; float* op;
    switch (rel) {
        case 0:  src = src_dd; dst = dst_dd; Fsrc = Fbase;                      op = out;                        break;
        case 1:  src = src_tt; dst = dst_tt; Fsrc = Fbase + 1ul * NNODES * DIM; op = out + (size_t)NNODES * DIM; break;
        case 2:  src = src_dt; dst = dst_dt; Fsrc = Fbase + 2ul * NNODES * DIM; op = out + (size_t)NNODES * DIM; break;
        default: src = src_td; dst = dst_td; Fsrc = Fbase + 3ul * NNODES * DIM; op = out;                        break;
    }
    const int wave = threadIdx.x >> 5;
    const int lane = threadIdx.x & 31;
    const size_t e = (size_t)blockIdx.x * 8 + wave;
    if (e >= NEDGE) return;

    const int s = src[e];
    const int d = dst[e];
    const float alpha = elog[(size_t)rel * NEDGE + e] / ssum[rel * NNODES + d];

    const float4 f = reinterpret_cast<const float4*>(Fsrc + (size_t)s * DIM)[lane];
    float* ob = op + (size_t)d * DIM + lane * 4;
    atomicAdd(ob + 0, alpha * f.x);
    atomicAdd(ob + 1, alpha * f.y);
    atomicAdd(ob + 2, alpha * f.z);
    atomicAdd(ob + 3, alpha * f.w);
}

// ---------------------------------------------------------------------------
extern "C" void kernel_launch(void* const* d_in, const int* in_sizes, int n_in,
                              void* d_out, int out_size, void* d_ws, size_t ws_size,
                              hipStream_t stream) {
    const float* x_d   = (const float*)d_in[0];
    const float* x_t   = (const float*)d_in[1];
    const float* W_dd  = (const float*)d_in[2];
    const float* W_tt  = (const float*)d_in[3];
    const float* W_dt  = (const float*)d_in[4];
    const float* al_dd = (const float*)d_in[5];
    const float* ar_dd = (const float*)d_in[6];
    const float* al_tt = (const float*)d_in[7];
    const float* ar_tt = (const float*)d_in[8];
    const float* al_dt = (const float*)d_in[9];
    const float* ar_dt = (const float*)d_in[10];
    const float* b_dd  = (const float*)d_in[11];
    const float* b_tt  = (const float*)d_in[12];
    const float* b_dt  = (const float*)d_in[13];
    const int* src_dd  = (const int*)d_in[14];
    const int* dst_dd  = (const int*)d_in[15];
    const int* src_tt  = (const int*)d_in[16];
    const int* dst_tt  = (const int*)d_in[17];
    const int* src_dt  = (const int*)d_in[18];
    const int* dst_dt  = (const int*)d_in[19];
    const int* src_td  = (const int*)d_in[20];
    const int* dst_td  = (const int*)d_in[21];
    float* out = (float*)d_out;

    // Workspace layout (~114 MB): F | el | er | ssum | elog | maxenc
    float* F      = (float*)d_ws;                 // 4*N*D f32
    float* el_arr = F + 4ul * NNODES * DIM;       // 4*N
    float* er_arr = el_arr + 4ul * NNODES;        // 4*N
    float* ssum   = er_arr + 4ul * NNODES;        // 4*N
    float* elog   = ssum + 4ul * NNODES;          // 4*E
    int*   maxenc = (int*)(elog + 4ul * NEDGE);   // 4*N

    {
        size_t total = 2ul * NNODES * DIM;
        int blocks = (int)((total + 255) / 256);
        init_kernel<<<blocks, 256, 0, stream>>>(out, b_dd, b_tt, b_dt, maxenc, ssum);
    }
    {
        int gb = (NNODES + 127) / 128;
        gemm_wmma<<<gb, 256, 0, stream>>>(x_d, W_dd, F,                       NNODES);
        gemm_wmma<<<gb, 256, 0, stream>>>(x_t, W_tt, F + 1ul * NNODES * DIM, NNODES);
        gemm_wmma<<<gb, 256, 0, stream>>>(x_d, W_dt, F + 2ul * NNODES * DIM, NNODES);
        gemm_wmma<<<gb, 256, 0, stream>>>(x_t, W_dt, F + 3ul * NNODES * DIM, NNODES);
    }
    dots_kernel<<<dim3((NNODES + 7) / 8, 4), 256, 0, stream>>>(
        F, al_dd, ar_dd, al_tt, ar_tt, al_dt, ar_dt, el_arr, er_arr);
    logits_kernel<<<dim3((NEDGE + 255) / 256, 4), 256, 0, stream>>>(
        el_arr, er_arr, src_dd, dst_dd, src_tt, dst_tt,
        src_dt, dst_dt, src_td, dst_td, elog, maxenc);
    expsum_kernel<<<dim3((NEDGE + 255) / 256, 4), 256, 0, stream>>>(
        dst_dd, dst_tt, dst_dt, dst_td, elog, maxenc, ssum);
    aggregate_kernel<<<dim3((NEDGE + 7) / 8, 4), 256, 0, stream>>>(
        F, elog, ssum, src_dd, dst_dd, src_tt, dst_tt,
        src_dt, dst_dt, src_td, dst_td, out);
}
